// EdgeLevelRNN_50465865728373
// MI455X (gfx1250) — matrix operand order repaired and checked by reading, not compiled, not run on hardware
//
#include <hip/hip_runtime.h>

typedef __attribute__((ext_vector_type(16))) _Float16 v16h;
typedef __attribute__((ext_vector_type(8)))  _Float16 v8h;
typedef __attribute__((ext_vector_type(8)))  float    v8f;

// ---------------------------------------------------------------------------
// Wave-level GEMM tile: C[16 x 16*ntiles] = A[16 x 32*Kt] * Wpk + bias
// A: f16 in LDS, row-major with stride `astride` (halves), column offset acol.
// Wpk: weights pre-packed in B-fragment order:
//      Wpk[((nt*Kt + kt)*32 + lane)*16 + j]  (lane 0-15: N=lane, K=0..15;
//                                             lane 16-31: N=lane-16, K=16..31)
// mode: 0 = store f32, 1 = add into f32, 2 = store f16, 3 = store f16 relu
// ---------------------------------------------------------------------------
__device__ __forceinline__ void wave_gemm(
    const _Float16* A, int astride, int acol,
    const _Float16* Wpk, const float* bias,
    int Kt, int ntStart, int ntEnd,
    void* out, int ostride, int ocol,
    int mode, int lane, int wave, int nwaves)
{
  const int m16 = lane & 15;
  const int sel = lane >> 4;
  for (int nt = ntStart + wave; nt < ntEnd; nt += nwaves) {
    float bv = bias ? bias[nt * 16 + m16] : 0.0f;
    v8f c;
#pragma unroll
    for (int r = 0; r < 8; ++r) c[r] = bv;
    for (int kt = 0; kt < Kt; ++kt) {
      // A fragment: contiguous K = kt*32 + sel*16 .. +15 (matches packed B)
      const _Float16* ap = A + m16 * astride + acol + kt * 32 + sel * 16;
      v8h lo = *(const v8h*)ap;            // K = kt*32 + sel*16 .. +7
      v8h hi = *(const v8h*)(ap + 8);      // K = kt*32 + sel*16 + 8 .. +15
      v16h afrag = __builtin_shufflevector(lo, hi,
          0, 1, 2, 3, 4, 5, 6, 7, 8, 9, 10, 11, 12, 13, 14, 15);
      // B fragment: one contiguous 32B load per lane (pre-packed)
      v16h bfrag = *(const v16h*)(Wpk + ((size_t)(nt * Kt + kt) * 32 + lane) * 16);
      c = __builtin_amdgcn_wmma_f32_16x16x32_f16(
          false, afrag, false, bfrag, (short)0, c, false, false);
    }
#pragma unroll
    for (int r = 0; r < 8; ++r) {
      int row = r + sel * 8;                           // C layout: M = r + sel*8
      int idx = row * ostride + ocol + nt * 16 + m16;  //           N = nt*16 + lane%16
      if (mode == 0)       ((float*)out)[idx]  = c[r];
      else if (mode == 1)  ((float*)out)[idx] += c[r];
      else if (mode == 2)  ((_Float16*)out)[idx] = (_Float16)c[r];
      else                 ((_Float16*)out)[idx] = (_Float16)fmaxf(c[r], 0.0f);
    }
  }
}

// ---------------------------------------------------------------------------
// Pack a row-major f32 weight W[O][I] (used as out = act @ W^T) into
// f16 B-fragment order for v_wmma_f32_16x16x32_f16.
// ---------------------------------------------------------------------------
__global__ void pack_weight(const float* __restrict__ src, _Float16* __restrict__ dst,
                            int O, int I) {
  int total = O * I;
  for (int e = blockIdx.x * blockDim.x + threadIdx.x; e < total;
       e += gridDim.x * blockDim.x) {
    int o = e / I, k = e % I;
    int nt = o >> 4, kt = k >> 5;
    int lane = ((k >> 4) & 1) * 16 + (o & 15);
    int j = k & 15;
    dst[(((size_t)(nt * (I >> 5) + kt)) * 32 + lane) * 16 + j] = (_Float16)src[e];
  }
}

// ---------------------------------------------------------------------------
// K/V precompute (time-invariant): K/V = prev @ Wk^T/Wv^T + b, stored f16 as
// [B][NH=4][N=64][hd=64]. One WG of 4 waves per 16 rows of flattened [B*N].
// ---------------------------------------------------------------------------
__global__ __launch_bounds__(128) void kv_precompute(
    const float* __restrict__ prev, const float* __restrict__ bk,
    const float* __restrict__ bv,
    const _Float16* __restrict__ WkPk, const _Float16* __restrict__ WvPk,
    _Float16* __restrict__ Kf16, _Float16* __restrict__ Vf16)
{
  __shared__ _Float16 a16[16 * 256];
  __shared__ _Float16 kls[16 * 256];
  __shared__ _Float16 vls[16 * 256];
  int tid = threadIdx.x, lane = tid & 31, wave = tid >> 5;
  size_t r0 = (size_t)blockIdx.x * 16;
  for (int idx = tid; idx < 4096; idx += 128) {
    int m = idx >> 8, e = idx & 255;
    a16[idx] = (_Float16)prev[(r0 + m) * 256 + e];
  }
  __syncthreads();
  wave_gemm(a16, 256, 0, WkPk, bk, 8, 0, 16, kls, 256, 0, 2, lane, wave, 4);
  wave_gemm(a16, 256, 0, WvPk, bv, 8, 0, 16, vls, 256, 0, 2, lane, wave, 4);
  __syncthreads();
  for (int idx = tid; idx < 4096; idx += 128) {
    int m = idx >> 8, e = idx & 255;
    size_t row = r0 + m;                // = b*64 + node
    size_t b = row >> 6;
    int node = (int)(row & 63);
    int h = e >> 6, d = e & 63;
    size_t off = ((b * 4 + h) * 64 + node) * 64 + d;
    Kf16[off] = kls[idx];
    Vf16[off] = vls[idx];
  }
}

__device__ __forceinline__ float sigmoidf_(float x) { return 1.0f / (1.0f + expf(-x)); }

// ---------------------------------------------------------------------------
// Persistent scan kernel: each WG (4 waves) owns 16 batch rows for all 64
// steps. Hidden state / activations live in LDS; weights stream from L2.
// ---------------------------------------------------------------------------
__global__ __launch_bounds__(128) void edge_rnn_scan(
    const float* __restrict__ x, const float* __restrict__ h_init,
    const float* __restrict__ lin_in_w, const float* __restrict__ lin_in_b,
    const float* __restrict__ b_ih, const float* __restrict__ b_hh,
    const float* __restrict__ bq, const float* __restrict__ bo,
    const float* __restrict__ comb_b, const float* __restrict__ out1_b,
    const float* __restrict__ out2_w, const float* __restrict__ out2_b,
    const _Float16* __restrict__ WihPk, const _Float16* __restrict__ WhhPk,
    const _Float16* __restrict__ WqPk, const _Float16* __restrict__ WoPk,
    const _Float16* __restrict__ WcombPk, const _Float16* __restrict__ Wout1Pk,
    const _Float16* __restrict__ Kf16, const _Float16* __restrict__ Vf16,
    float* __restrict__ out)
{
  extern __shared__ unsigned char smem_raw[];
  float* G  = (float*)smem_raw;          // [16][768] gx (then gx+gh for r,z)
  float* Hn = G + 16 * 768;              // [16][256] gh n-gate
  float* sc = Hn + 16 * 256;             // [16][4][64] scores/probs
  _Float16* hbuf    = (_Float16*)(sc + 16 * 4 * 64); // [2][16][256] hidden
  _Float16* inpA    = hbuf + 2 * 4096;   // [16][256] embedded input
  _Float16* comb    = inpA + 4096;       // [16][512] [h_top | attn_out]
  _Float16* qbuf    = comb + 8192;       // [16][256]
  _Float16* attnv   = qbuf + 4096;       // [16][256]
  _Float16* projbuf = attnv + 4096;      // [16][256]
  _Float16* o1buf   = projbuf + 4096;    // [16][256]

  const int tid = threadIdx.x, lane = tid & 31, wave = tid >> 5;
  const int NW = 4;
  const int b0 = blockIdx.x * 16;

  // init hidden: layer0 = h_init, layer1 = 0
  for (int idx = tid; idx < 4096; idx += 128) {
    int m = idx >> 8, e = idx & 255;
    hbuf[idx]        = (_Float16)h_init[(size_t)(b0 + m) * 256 + e];
    hbuf[4096 + idx] = (_Float16)0.0f;
  }
  __syncthreads();

  for (int t = 0; t < 64; ++t) {
    // embedded_x = relu(x * lin_in_w + b), F == 1
    for (int idx = tid; idx < 4096; idx += 128) {
      int m = idx >> 8, e = idx & 255;
      float xv = x[(size_t)(b0 + m) * 64 + t];
      inpA[idx] = (_Float16)fmaxf(xv * lin_in_w[e] + lin_in_b[e], 0.0f);
    }
    __syncthreads();

    // ---- stacked GRU (PyTorch r,z,n gate order) ----
    for (int l = 0; l < 2; ++l) {
      const _Float16* cur = (l == 0) ? inpA : hbuf;   // input to layer l
      _Float16* hl = hbuf + l * 4096;                 // old/new hidden of layer l
      const _Float16* Wih = WihPk + (size_t)l * 196608;
      const _Float16* Whh = WhhPk + (size_t)l * 196608;
      // gx = cur @ Wih^T + b_ih  -> G[16][768]
      wave_gemm(cur, 256, 0, Wih, b_ih + l * 768, 8, 0, 48, G, 768, 0, 0, lane, wave, NW);
      // gh r,z: add into G (same nt range handled by same wave -> no barrier)
      wave_gemm(hl, 256, 0, Whh, b_hh + l * 768, 8, 0, 32, G, 768, 0, 1, lane, wave, NW);
      // gh n: keep separate for tanh(gx_n + r*gh_n)
      wave_gemm(hl, 256, 0, Whh, b_hh + l * 768, 8, 32, 48, Hn, 256, -512, 0, lane, wave, NW);
      __syncthreads();
      for (int idx = tid; idx < 4096; idx += 128) {
        int m = idx >> 8, e = idx & 255;
        float r = sigmoidf_(G[m * 768 + e]);
        float z = sigmoidf_(G[m * 768 + 256 + e]);
        float n = tanhf(G[m * 768 + 512 + e] + r * Hn[m * 256 + e]);
        float hv = (1.0f - z) * n + z * (float)hl[idx];
        hl[idx] = (_Float16)hv;
        if (l == 1) comb[m * 512 + e] = (_Float16)hv;  // left half of combined
      }
      __syncthreads();
    }

    // ---- cross attention: q = h_top @ wq^T + bq ----
    wave_gemm(hbuf + 4096, 256, 0, WqPk, bq, 8, 0, 16, qbuf, 256, 0, 2, lane, wave, NW);
    __syncthreads();

    // scores (batched per-row matvec; K/V are per batch element -> VALU,
    // but with 16B-vector loads on the dominant K stream)
    const float scale = 0.125f;  // 1/sqrt(64)
    for (int idx = tid; idx < 4096; idx += 128) {
      int m = idx >> 8, h = (idx >> 6) & 3, n = idx & 63;
      const _Float16* qp = qbuf + m * 256 + h * 64;
      const _Float16* kp = Kf16 + (((size_t)(b0 + m) * 4 + h) * 64 + n) * 64;
      float s = 0.0f;
#pragma unroll
      for (int dc = 0; dc < 8; ++dc) {
        v8h qv = *(const v8h*)(qp + dc * 8);   // ds_load_b128
        v8h kv = *(const v8h*)(kp + dc * 8);   // global_load_b128
#pragma unroll
        for (int j = 0; j < 8; ++j) s += (float)qv[j] * (float)kv[j];
      }
      sc[idx] = s * scale;
    }
    __syncthreads();
    if (tid < 64) {  // softmax over 64 nodes per (row, head)
      float* p = sc + tid * 64;
      float mx = p[0];
      for (int n = 1; n < 64; ++n) mx = fmaxf(mx, p[n]);
      float sum = 0.0f;
      for (int n = 0; n < 64; ++n) { float e2 = expf(p[n] - mx); p[n] = e2; sum += e2; }
      float inv = 1.0f / sum;
      for (int n = 0; n < 64; ++n) p[n] *= inv;
    }
    __syncthreads();
    // attn: tasks = (m, h, 8-wide d block) so V reads are contiguous 16B
    for (int task = tid; task < 512; task += 128) {
      int m = task >> 5, h = (task >> 3) & 3, db = task & 7;
      const float* pp = sc + m * 256 + h * 64;
      const _Float16* vbase = Vf16 + ((size_t)(b0 + m) * 4 + h) * 4096 + db * 8;
      float acc[8];
#pragma unroll
      for (int j = 0; j < 8; ++j) acc[j] = 0.0f;
      for (int n = 0; n < 64; ++n) {
        v8h vv = *(const v8h*)(vbase + n * 64);  // global_load_b128
        float p = pp[n];
#pragma unroll
        for (int j = 0; j < 8; ++j) acc[j] += p * (float)vv[j];
      }
#pragma unroll
      for (int j = 0; j < 8; ++j)
        attnv[m * 256 + h * 64 + db * 8 + j] = (_Float16)acc[j];
    }
    __syncthreads();

    // attn_out -> right half of combined
    wave_gemm(attnv, 256, 0, WoPk, bo, 8, 0, 16, comb, 512, 256, 2, lane, wave, NW);
    __syncthreads();
    // proj = relu(combined @ comb_w^T + comb_b)
    wave_gemm(comb, 512, 0, WcombPk, comb_b, 16, 0, 16, projbuf, 256, 0, 3, lane, wave, NW);
    __syncthreads();
    // o1 = relu(proj @ out1_w^T + out1_b)
    wave_gemm(projbuf, 256, 0, Wout1Pk, out1_b, 8, 0, 16, o1buf, 256, 0, 3, lane, wave, NW);
    __syncthreads();
    // o2 = o1 @ out2_w^T + out2_b; out = sigmoid(o2)   (F == 1)
    if (tid < 16) {
      float s = out2_b[0];
      const _Float16* op = o1buf + tid * 256;
#pragma unroll
      for (int ec = 0; ec < 32; ++ec) {
        v8h ov = *(const v8h*)(op + ec * 8);
#pragma unroll
        for (int j = 0; j < 8; ++j) s += (float)ov[j] * out2_w[ec * 8 + j];
      }
      out[(size_t)(b0 + tid) * 64 + t] = sigmoidf_(s);
    }
    __syncthreads();
  }
}

// ---------------------------------------------------------------------------
extern "C" void kernel_launch(void* const* d_in, const int* in_sizes, int n_in,
                              void* d_out, int out_size, void* d_ws, size_t ws_size,
                              hipStream_t stream) {
  const float* x     = (const float*)d_in[0];
  const float* prev  = (const float*)d_in[1];
  const float* hinit = (const float*)d_in[2];
  const float* linw  = (const float*)d_in[3];
  const float* linb  = (const float*)d_in[4];
  const float* gih   = (const float*)d_in[5];
  const float* ghh   = (const float*)d_in[6];
  const float* bih   = (const float*)d_in[7];
  const float* bhh   = (const float*)d_in[8];
  const float* wq    = (const float*)d_in[9];
  const float* bq    = (const float*)d_in[10];
  const float* wk    = (const float*)d_in[11];
  const float* bk    = (const float*)d_in[12];
  const float* wv    = (const float*)d_in[13];
  const float* bv    = (const float*)d_in[14];
  const float* wo    = (const float*)d_in[15];
  const float* bo    = (const float*)d_in[16];
  const float* combw = (const float*)d_in[17];
  const float* combb = (const float*)d_in[18];
  const float* o1w   = (const float*)d_in[19];
  const float* o1b   = (const float*)d_in[20];
  const float* o2w   = (const float*)d_in[21];
  const float* o2b   = (const float*)d_in[22];
  float* out = (float*)d_out;

  // workspace layout (halves)
  _Float16* ws = (_Float16*)d_ws;
  size_t off = 0;
  _Float16* WihPk   = ws + off; off += 2 * 768 * 256;        // 393216
  _Float16* WhhPk   = ws + off; off += 2 * 768 * 256;
  _Float16* WqPk    = ws + off; off += 256 * 256;
  _Float16* WkPk    = ws + off; off += 256 * 256;
  _Float16* WvPk    = ws + off; off += 256 * 256;
  _Float16* WoPk    = ws + off; off += 256 * 256;
  _Float16* WcombPk = ws + off; off += 256 * 512;
  _Float16* Wout1Pk = ws + off; off += 256 * 256;
  _Float16* Kf16    = ws + off; off += (size_t)2048 * 4 * 64 * 64;
  _Float16* Vf16    = ws + off; off += (size_t)2048 * 4 * 64 * 64;

  // pack all GEMM weights into WMMA B-fragment order (f16)
  pack_weight<<<768, 256, 0, stream>>>(gih,            WihPk,            768, 256);
  pack_weight<<<768, 256, 0, stream>>>(gih + 196608,   WihPk + 196608,   768, 256);
  pack_weight<<<768, 256, 0, stream>>>(ghh,            WhhPk,            768, 256);
  pack_weight<<<768, 256, 0, stream>>>(ghh + 196608,   WhhPk + 196608,   768, 256);
  pack_weight<<<256, 256, 0, stream>>>(wq,    WqPk,    256, 256);
  pack_weight<<<256, 256, 0, stream>>>(wk,    WkPk,    256, 256);
  pack_weight<<<256, 256, 0, stream>>>(wv,    WvPk,    256, 256);
  pack_weight<<<256, 256, 0, stream>>>(wo,    WoPk,    256, 256);
  pack_weight<<<512, 256, 0, stream>>>(combw, WcombPk, 256, 512);
  pack_weight<<<256, 256, 0, stream>>>(o1w,   Wout1Pk, 256, 256);

  // time-invariant K/V projections: [B*N]/16 = 8192 tiles
  kv_precompute<<<8192, 128, 0, stream>>>(prev, bk, bv, WkPk, WvPk, Kf16, Vf16);

  // persistent scan: 2048/16 = 128 batch tiles, 4 waves each
  size_t smemBytes = (16 * 768 + 16 * 256 + 16 * 4 * 64) * sizeof(float) +
                     (2 * 4096 + 4096 + 8192 + 4096 + 4096 + 4096 + 4096) * sizeof(_Float16);
  edge_rnn_scan<<<128, 128, smemBytes, stream>>>(
      x, hinit, linw, linb, bih, bhh, bq, bo, combb, o1b, o2w, o2b,
      WihPk, WhhPk, WqPk, WoPk, WcombPk, Wout1Pk, Kf16, Vf16, out);
}
